// ChatGLM3DecoderLayer_11467562680664
// MI455X (gfx1250) — matrix operand order repaired and checked by reading, not compiled, not run on hardware
//
#include <hip/hip_runtime.h>

// ---------------- problem constants ----------------
constexpr int Bc   = 2;
constexpr int Sc   = 1024;
constexpr int Hc   = 4096;
constexpr int NHc  = 32;
constexpr int NKVc = 2;
constexpr int Dc   = 128;
constexpr int FFNc = 13696;
constexpr int Mrows = Bc * Sc;               // 2048
constexpr int NQKV  = (NHc + 2 * NKVc) * Dc; // 4608

// ---------------- bf16 helpers ----------------
typedef __attribute__((ext_vector_type(16))) __bf16 bf16x16;
typedef __attribute__((ext_vector_type(8)))  float  f32x8;
typedef int v4i __attribute__((vector_size(16)));

union Frag { unsigned int u[8]; bf16x16 v; };

__device__ __forceinline__ unsigned short f2bf(float f) {
  unsigned int u = __builtin_bit_cast(unsigned int, f);
  u += 0x7FFFu + ((u >> 16) & 1u);           // round-to-nearest-even
  return (unsigned short)(u >> 16);
}
__device__ __forceinline__ float bf2f(unsigned short h) {
  unsigned int u = ((unsigned int)h) << 16;
  return __builtin_bit_cast(float, u);
}
__device__ __forceinline__ unsigned int pack2(float a, float b) {
  return (unsigned int)f2bf(a) | ((unsigned int)f2bf(b) << 16);
}

// ---------------- CDNA5 async global->LDS copy (ASYNCcnt path) ----------------
#if defined(__AMDGCN__) && __has_builtin(__builtin_amdgcn_global_load_async_to_lds_b128) && \
    __has_builtin(__builtin_amdgcn_s_wait_asynccnt)
#define USE_ASYNC_LDS 1
#else
#define USE_ASYNC_LDS 0
#endif

__device__ __forceinline__ void async_cp16(const unsigned short* src,
                                           unsigned short* lds_dst) {
#if USE_ASYNC_LDS
  __builtin_amdgcn_global_load_async_to_lds_b128(
      (__attribute__((address_space(1))) v4i*)(void*)const_cast<unsigned short*>(src),
      (__attribute__((address_space(3))) v4i*)(void*)lds_dst, 0, 0);
#else
  *(uint4*)lds_dst = *(const uint4*)src;
#endif
}
__device__ __forceinline__ void async_wait_all() {
#if USE_ASYNC_LDS
  __builtin_amdgcn_s_wait_asynccnt(0);
#endif
}

// ---------------- fp32 -> bf16 weight convert ----------------
__global__ __launch_bounds__(256) void cvt_f32_bf16(const float* __restrict__ src,
                                                    unsigned short* __restrict__ dst,
                                                    size_t n) {
  size_t i = (size_t)blockIdx.x * blockDim.x + threadIdx.x;
  size_t stride = (size_t)gridDim.x * blockDim.x;
  for (; i < n; i += stride) dst[i] = f2bf(src[i]);
}

// ---------------- RMSNorm (fp32 in, bf16 out) ----------------
__global__ __launch_bounds__(256) void rmsnorm_bf16(const float* __restrict__ x,
                                                    const float* __restrict__ w,
                                                    unsigned short* __restrict__ out) {
  __shared__ float red[256];
  const int row = blockIdx.x;
  const float* xr = x + (size_t)row * Hc;
  float s = 0.f;
#pragma unroll
  for (int i = 0; i < Hc / 256; ++i) {
    float v = xr[threadIdx.x + i * 256];
    s += v * v;
  }
  red[threadIdx.x] = s;
  __syncthreads();
  for (int off = 128; off > 0; off >>= 1) {
    if (threadIdx.x < off) red[threadIdx.x] += red[threadIdx.x + off];
    __syncthreads();
  }
  float rs = rsqrtf(red[0] / (float)Hc + 1e-5f);
#pragma unroll
  for (int i = 0; i < Hc / 256; ++i) {
    int c = threadIdx.x + i * 256;
    out[(size_t)row * Hc + c] = f2bf(xr[c] * rs * w[c]);
  }
}

// ---------------- bf16 WMMA GEMM: C[M,N] = A[M,K] * W[N,K]^T ----------------
// block tile 128x128, BK=32, 256 threads = 8 waves in 4(M) x 2(N) grid,
// each wave: 2x4 accumulator tiles of 16x16.
// Double-buffered LDS staged with GLOBAL_LOAD_ASYNC_TO_LDS_B128.
#define BM 128
#define BN 128
#define BK 32

__global__ __launch_bounds__(256) void gemm_bf16(const unsigned short* __restrict__ A,
                                                 const unsigned short* __restrict__ W,
                                                 float* __restrict__ outF,
                                                 unsigned short* __restrict__ outB,
                                                 const float* __restrict__ bias,
                                                 const float* __restrict__ res,
                                                 int M, int N, int K) {
  __shared__ __align__(16) unsigned short As[2][BM * BK];
  __shared__ __align__(16) unsigned short Bs[2][BN * BK];

  const int tid  = threadIdx.x;
  const int lane = tid & 31;
  const int wv   = tid >> 5;
  const int wm   = wv >> 1;           // 0..3
  const int wn   = wv & 1;            // 0..1
  const int lm   = lane & 15;
  const int hi   = lane >> 4;
  const int bm   = blockIdx.y * BM;
  const int bn   = blockIdx.x * BN;

  // per-thread staging slots (2 x 16B per array)
  const int r0 = tid >> 2;
  const int c0 = (tid & 3) * 8;
  const int r1 = (tid + 256) >> 2;
  const int c1 = ((tid + 256) & 3) * 8;

  f32x8 acc[2][4];
#pragma unroll
  for (int mt = 0; mt < 2; ++mt)
#pragma unroll
    for (int nt = 0; nt < 4; ++nt)
      acc[mt][nt] = f32x8{0.f, 0.f, 0.f, 0.f, 0.f, 0.f, 0.f, 0.f};

  auto stage = [&](int k0, int buf) {
    async_cp16(A + (size_t)(bm + r0) * K + k0 + c0, &As[buf][r0 * BK + c0]);
    async_cp16(W + (size_t)(bn + r0) * K + k0 + c0, &Bs[buf][r0 * BK + c0]);
    async_cp16(A + (size_t)(bm + r1) * K + k0 + c1, &As[buf][r1 * BK + c1]);
    async_cp16(W + (size_t)(bn + r1) * K + k0 + c1, &Bs[buf][r1 * BK + c1]);
  };

  stage(0, 0);
  int cur = 0;

  for (int k0 = 0; k0 < K; k0 += BK) {
    async_wait_all();     // this wave's staged tile is in LDS
    __syncthreads();      // every wave's share is in LDS
    if (k0 + BK < K) stage(k0 + BK, cur ^ 1);  // DMA next tile during compute

    const unsigned short* as = As[cur];
    const unsigned short* bs = Bs[cur];

    Frag af[2], bf_[4];
#pragma unroll
    for (int mt = 0; mt < 2; ++mt) {
      int m = wm * 32 + mt * 16 + lm;
#pragma unroll
      for (int v = 0; v < 8; ++v) {
        int k = ((v < 4) ? 0 : 16) + hi * 8 + (v & 3) * 2; // A 16x32 layout
        af[mt].u[v] = *(const unsigned int*)&as[m * BK + k];
      }
    }
#pragma unroll
    for (int nt = 0; nt < 4; ++nt) {
      int n = wn * 64 + nt * 16 + lm;
#pragma unroll
      for (int v = 0; v < 8; ++v) {
        int k = ((v < 4) ? 0 : 16) + hi * 8 + (v & 3) * 2; // B 32x16 layout
        bf_[nt].u[v] = *(const unsigned int*)&bs[n * BK + k];
      }
    }
#pragma unroll
    for (int mt = 0; mt < 2; ++mt)
#pragma unroll
      for (int nt = 0; nt < 4; ++nt)
        acc[mt][nt] = __builtin_amdgcn_wmma_f32_16x16x32_bf16(
            false, af[mt].v, false, bf_[nt].v, (short)0, acc[mt][nt], false, false);
    __syncthreads();      // all reads of buf `cur` done before it is re-targeted
    cur ^= 1;
  }

  // epilogue: +bias, +residual, store fp32 or bf16
#pragma unroll
  for (int mt = 0; mt < 2; ++mt) {
#pragma unroll
    for (int v = 0; v < 8; ++v) {
      int row = bm + wm * 32 + mt * 16 + v + 8 * hi;
#pragma unroll
      for (int nt = 0; nt < 4; ++nt) {
        int col = bn + wn * 64 + nt * 16 + lm;
        size_t o = (size_t)row * N + col;
        float val = acc[mt][nt][v];
        if (bias) val += bias[col];
        if (res)  val += res[o];
        if (outF) outF[o] = val;
        else      outB[o] = f2bf(val);
      }
    }
  }
}

// ---------------- RoPE in-place on fused QKV (q heads 0..31, k heads 32..33) ----------------
__global__ __launch_bounds__(256) void rope_inplace(float* __restrict__ qkv,
                                                    const int* __restrict__ positions) {
  int idx = blockIdx.x * 256 + threadIdx.x;
  const int total = Bc * Sc * (NHc + NKVc) * (Dc / 2);
  if (idx >= total) return;
  int i  = idx & 63;
  int h  = (idx >> 6) % (NHc + NKVc);
  int bs = idx / (64 * (NHc + NKVc));
  float pos = (float)positions[bs];
  float freq = __expf(-(float)i * (9.210340371976184f / 64.0f)); // ln(10000)
  float sn, cs;
  __sincosf(pos * freq, &sn, &cs);
  float* p = qkv + (size_t)bs * NQKV + h * Dc;
  float x1 = p[i], x2 = p[i + 64];
  p[i]      = x1 * cs - x2 * sn;
  p[i + 64] = x2 * cs + x1 * sn;
}

// ---------------- flash attention (GQA), bf16 WMMA, online softmax ----------------
// grid: (S/16, NH/8, B); block 256 = 8 waves; wave w handles head blockIdx.y*8+w.
// All 8 heads share the same KV group -> K/V staged once in LDS per chunk.
__global__ __launch_bounds__(256) void attention(const float* __restrict__ qkv,
                                                 unsigned short* __restrict__ attn) {
  __shared__ __align__(16) unsigned short Ks[32 * 128];   // [key][channel]
  __shared__ __align__(16) unsigned short Vs[128 * 32];   // [channel][key] (transposed)
  __shared__ __align__(16) unsigned short Ps[8][16 * 32]; // per-wave probs

  const int tid  = threadIdx.x;
  const int lane = tid & 31;
  const int wv   = tid >> 5;
  const int lm   = lane & 15;
  const int hi   = lane >> 4;
  const int qbase = blockIdx.x * 16;
  const int head  = blockIdx.y * 8 + wv;
  const int b     = blockIdx.z;
  const int g     = head >> 4;
  const float scale = 0.08838834764831845f; // 1/sqrt(128)

  // Q fragments (K-dim 128 -> 4 A-frags), scale folded in
  Frag qf[4];
#pragma unroll
  for (int cb4 = 0; cb4 < 4; ++cb4) {
    const float* qp = qkv + (size_t)(b * Sc + qbase + lm) * NQKV + head * Dc;
#pragma unroll
    for (int v = 0; v < 8; ++v) {
      int k = cb4 * 32 + ((v < 4) ? 0 : 16) + hi * 8 + (v & 3) * 2;
      qf[cb4].u[v] = pack2(qp[k] * scale, qp[k + 1] * scale);
    }
  }

  float mstate[8], lstate[8];
  f32x8 O[8];
#pragma unroll
  for (int v = 0; v < 8; ++v) { mstate[v] = -3.0e38f; lstate[v] = 0.f; }
#pragma unroll
  for (int nt = 0; nt < 8; ++nt)
    O[nt] = f32x8{0.f, 0.f, 0.f, 0.f, 0.f, 0.f, 0.f, 0.f};

  for (int jk = 0; jk < qbase + 16; jk += 32) {
    __syncthreads();
    // cooperative KV chunk load (fp32 -> bf16 conversion in flight)
#pragma unroll
    for (int i = 0; i < 2; ++i) {
      int t  = tid + i * 256;
      int kk = t >> 4;          // key 0..31
      int c  = (t & 15) * 8;    // channel 0..120
      const float* kp = qkv + (size_t)(b * Sc + jk + kk) * NQKV + NHc * Dc + g * Dc + c;
      const float* vp = qkv + (size_t)(b * Sc + jk + kk) * NQKV + (NHc + NKVc) * Dc + g * Dc + c;
      unsigned int* kd = (unsigned int*)&Ks[kk * 128 + c];
#pragma unroll
      for (int e = 0; e < 4; ++e) kd[e] = pack2(kp[2 * e], kp[2 * e + 1]);
#pragma unroll
      for (int e = 0; e < 8; ++e) Vs[(c + e) * 32 + kk] = f2bf(vp[e]);
    }
    __syncthreads();

    // scores: two 16x16 tiles over this 32-key chunk
    f32x8 s[2];
#pragma unroll
    for (int t2 = 0; t2 < 2; ++t2) {
      f32x8 c0 = f32x8{0.f, 0.f, 0.f, 0.f, 0.f, 0.f, 0.f, 0.f};
#pragma unroll
      for (int cb4 = 0; cb4 < 4; ++cb4) {
        Frag kb;
#pragma unroll
        for (int v = 0; v < 8; ++v) {
          int ch = cb4 * 32 + ((v < 4) ? 0 : 16) + hi * 8 + (v & 3) * 2;
          kb.u[v] = *(const unsigned int*)&Ks[(t2 * 16 + lm) * 128 + ch];
        }
        c0 = __builtin_amdgcn_wmma_f32_16x16x32_bf16(false, qf[cb4].v, false, kb.v,
                                                     (short)0, c0, false, false);
      }
      s[t2] = c0;
    }

    // online softmax per row (row = v + 8*hi), half-wave reductions
#pragma unroll
    for (int v = 0; v < 8; ++v) {
      int row  = v + 8 * hi;
      int qrow = qbase + row;
      float s0 = s[0][v]; if (jk + lm > qrow)       s0 = -3.0e38f;
      float s1 = s[1][v]; if (jk + 16 + lm > qrow)  s1 = -3.0e38f;
      float mx = fmaxf(s0, s1);
#pragma unroll
      for (int off = 1; off < 16; off <<= 1)
        mx = fmaxf(mx, __shfl_xor(mx, off, 32));
      float mnew = fmaxf(mstate[v], mx);
      float scl  = __expf(mstate[v] - mnew);
      float e0   = __expf(s0 - mnew);
      float e1   = __expf(s1 - mnew);
      float rs   = e0 + e1;
#pragma unroll
      for (int off = 1; off < 16; off <<= 1)
        rs += __shfl_xor(rs, off, 32);
      lstate[v] = lstate[v] * scl + rs;
      mstate[v] = mnew;
#pragma unroll
      for (int nt = 0; nt < 8; ++nt) O[nt][v] *= scl;
      Ps[wv][row * 32 + lm]      = f2bf(e0);
      Ps[wv][row * 32 + 16 + lm] = f2bf(e1);
    }

    // P(16x32 keys) x V(32x128) accumulated into O
    Frag pf;
#pragma unroll
    for (int v = 0; v < 8; ++v) {
      int k = ((v < 4) ? 0 : 16) + hi * 8 + (v & 3) * 2;
      pf.u[v] = *(const unsigned int*)&Ps[wv][lm * 32 + k];
    }
#pragma unroll
    for (int nt = 0; nt < 8; ++nt) {
      Frag vf;
#pragma unroll
      for (int v = 0; v < 8; ++v) {
        int key = ((v < 4) ? 0 : 16) + hi * 8 + (v & 3) * 2;
        vf.u[v] = *(const unsigned int*)&Vs[(nt * 16 + lm) * 32 + key];
      }
      O[nt] = __builtin_amdgcn_wmma_f32_16x16x32_bf16(false, pf.v, false, vf.v,
                                                      (short)0, O[nt], false, false);
    }
  }

  // normalize and store bf16 [B,S,H] (head-major channels)
#pragma unroll
  for (int v = 0; v < 8; ++v) {
    float invl = 1.0f / lstate[v];
    int srow = qbase + v + 8 * hi;
    size_t base = (size_t)(b * Sc + srow) * Hc + head * Dc + lm;
#pragma unroll
    for (int nt = 0; nt < 8; ++nt)
      attn[base + nt * 16] = f2bf(O[nt][v] * invl);
  }
}

// ---------------- SwiGLU: silu(a) * b, bf16 -> bf16 ----------------
__global__ __launch_bounds__(256) void swiglu_bf16(const unsigned short* __restrict__ inter,
                                                   unsigned short* __restrict__ act) {
  size_t idx = (size_t)blockIdx.x * 256 + threadIdx.x;
  const size_t total = (size_t)Mrows * FFNc;
  if (idx >= total) return;
  size_t r = idx / FFNc, c = idx % FFNc;
  float a  = bf2f(inter[r * (2 * FFNc) + c]);
  float bb = bf2f(inter[r * (2 * FFNc) + FFNc + c]);
  float sa = a / (1.0f + __expf(-a));
  act[r * FFNc + c] = f2bf(sa * bb);
}

// ---------------- launch ----------------
extern "C" void kernel_launch(void* const* d_in, const int* in_sizes, int n_in,
                              void* d_out, int out_size, void* d_ws, size_t ws_size,
                              hipStream_t stream) {
  const int*   positions     = (const int*)d_in[0];
  const float* hidden_states = (const float*)d_in[1];
  const float* ln1_w         = (const float*)d_in[2];
  const float* wqkv          = (const float*)d_in[3];
  const float* bqkv          = (const float*)d_in[4];
  const float* wo            = (const float*)d_in[5];
  const float* ln2_w         = (const float*)d_in[6];
  const float* w1            = (const float*)d_in[7];
  const float* w2            = (const float*)d_in[8];
  float* out = (float*)d_out;

  const size_t nWqkv = (size_t)NQKV * Hc;       // 18,874,368
  const size_t nWo   = (size_t)Hc * Hc;         // 16,777,216
  const size_t nW1   = (size_t)(2 * FFNc) * Hc; // 112,197,632
  const size_t nW2   = (size_t)Hc * FFNc;       // 56,098,816

  char* p = (char*)d_ws;
  auto carve = [&](size_t bytes) -> char* {
    char* r = p;
    p += (bytes + 255) & ~(size_t)255;
    return r;
  };
  unsigned short* wqkv_bf = (unsigned short*)carve(nWqkv * 2);
  unsigned short* wo_bf   = (unsigned short*)carve(nWo * 2);
  unsigned short* w1_bf   = (unsigned short*)carve(nW1 * 2);
  unsigned short* w2_bf   = (unsigned short*)carve(nW2 * 2);
  unsigned short* xn      = (unsigned short*)carve((size_t)Mrows * Hc * 2);
  float*          qkv     = (float*)carve((size_t)Mrows * NQKV * 4);
  unsigned short* attnb   = (unsigned short*)carve((size_t)Mrows * Hc * 2);
  float*          hidden  = (float*)carve((size_t)Mrows * Hc * 4);
  unsigned short* yb      = (unsigned short*)carve((size_t)Mrows * Hc * 2);
  unsigned short* interb  = (unsigned short*)carve((size_t)Mrows * 2 * FFNc * 2);
  unsigned short* actb    = (unsigned short*)carve((size_t)Mrows * FFNc * 2);

  // 1) weights fp32 -> bf16 (once per launch)
  cvt_f32_bf16<<<4096, 256, 0, stream>>>(wqkv, wqkv_bf, nWqkv);
  cvt_f32_bf16<<<4096, 256, 0, stream>>>(wo,   wo_bf,   nWo);
  cvt_f32_bf16<<<8192, 256, 0, stream>>>(w1,   w1_bf,   nW1);
  cvt_f32_bf16<<<8192, 256, 0, stream>>>(w2,   w2_bf,   nW2);

  // 2) RMSNorm 1 -> xn (bf16)
  rmsnorm_bf16<<<Mrows, 256, 0, stream>>>(hidden_states, ln1_w, xn);

  // 3) fused QKV projection (+bias) -> qkv fp32
  gemm_bf16<<<dim3(NQKV / BN, Mrows / BM), 256, 0, stream>>>(
      xn, wqkv_bf, qkv, nullptr, bqkv, nullptr, Mrows, NQKV, Hc);

  // 4) RoPE in place on q,k
  {
    int total = Bc * Sc * (NHc + NKVc) * (Dc / 2);
    rope_inplace<<<(total + 255) / 256, 256, 0, stream>>>(qkv, positions);
  }

  // 5) flash attention -> attnb (bf16)
  attention<<<dim3(Sc / 16, NHc / 8, Bc), 256, 0, stream>>>(qkv, attnb);

  // 6) output projection + residual -> hidden fp32
  gemm_bf16<<<dim3(Hc / BN, Mrows / BM), 256, 0, stream>>>(
      attnb, wo_bf, hidden, nullptr, nullptr, hidden_states, Mrows, Hc, Hc);

  // 7) RMSNorm 2 -> yb (bf16)
  rmsnorm_bf16<<<Mrows, 256, 0, stream>>>(hidden, ln2_w, yb);

  // 8) h -> 4h (stores bf16 directly)
  gemm_bf16<<<dim3(2 * FFNc / BN, Mrows / BM), 256, 0, stream>>>(
      yb, w1_bf, nullptr, interb, nullptr, nullptr, Mrows, 2 * FFNc, Hc);

  // 9) SwiGLU -> actb (bf16)
  {
    size_t total = (size_t)Mrows * FFNc;
    swiglu_bf16<<<(unsigned)((total + 255) / 256), 256, 0, stream>>>(interb, actb);
  }

  // 10) 4h -> h + residual -> d_out fp32
  gemm_bf16<<<dim3(Hc / BN, Mrows / BM), 256, 0, stream>>>(
      actb, w2_bf, out, nullptr, nullptr, hidden, Mrows, Hc, FFNc);
}
